// SelfAttention_74698071212118
// MI455X (gfx1250) — matrix operand order repaired and checked by reading, not compiled, not run on hardware
//
#include <hip/hip_runtime.h>

typedef __attribute__((ext_vector_type(16))) _Float16 v16h;
typedef __attribute__((ext_vector_type(8)))  _Float16 v8h;
typedef __attribute__((ext_vector_type(4)))  _Float16 v4h;
typedef __attribute__((ext_vector_type(8)))  float    v8f;

#define DIN  512
#define DNEW 512
#define SEQ  2048
#define NB   8

union V16 { v16h v; v8h h[2]; };
union V8H { v8h v; _Float16 e[8]; };

__device__ __forceinline__ float lane_bcast(float x, int srcLane) {
    int r = __builtin_amdgcn_ds_bpermute(srcLane << 2, __builtin_bit_cast(int, x));
    return __builtin_bit_cast(float, r);
}
__device__ __forceinline__ float xor16(float x) {
    // group-of-32 swizzle: xor_mask=0x10, and_mask=0x1f -> offset 0x401F
    int r = __builtin_amdgcn_ds_swizzle(__builtin_bit_cast(int, x), 0x401F);
    return __builtin_bit_cast(float, r);
}

// -----------------------------------------------------------------------------
// Kernel 1: QKV projection, fp32 in -> f16 out.
//   z==0: Q = (X*Wq + bq) * rsqrt(512) * log2(e)   (row-major [s][e])
//   z==1: K = X*Wk + bk                            (row-major [s][e])
//   z==2: Vt = (X*Wv + bv)^T                       (transposed [e][s])
// 64x64 tile per workgroup, 8 waves (4 M-groups x 2 N-groups), K-step 32.
// -----------------------------------------------------------------------------
__launch_bounds__(256)
__global__ void qkv_proj_kernel(const float* __restrict__ X,
                                const float* __restrict__ Wq, const float* __restrict__ bq,
                                const float* __restrict__ Wk, const float* __restrict__ bk,
                                const float* __restrict__ Wv, const float* __restrict__ bv,
                                _Float16* __restrict__ Q, _Float16* __restrict__ K,
                                _Float16* __restrict__ Vt)
{
    const int z = blockIdx.z;
    const float* W    = (z == 0) ? Wq : (z == 1) ? Wk : Wv;
    const float* bias = (z == 0) ? bq : (z == 1) ? bk : bv;
    // fold softmax scale and log2(e) into Q so attention can use exp2
    const float scale = (z == 0) ? (0.04419417382415922f * 1.4426950408889634f) : 1.0f;

    const int b  = blockIdx.x >> 5;            // 32 s-tiles per batch
    const int s0 = (blockIdx.x & 31) * 64;
    const int n0 = blockIdx.y * 64;

    __shared__ _Float16 Xs[64 * 40];           // [m][k], pitch 40
    __shared__ _Float16 Wts[64 * 40];          // [n][k], pitch 40 (W tile transposed)
    __shared__ _Float16 Cs[64 * 72];           // output tile staging, pitch 72

    const int tid  = threadIdx.x;
    const int lane = tid & 31;
    const int wave = tid >> 5;
    const int m0   = (wave >> 1) * 16;         // wave row group
    const int n0w  = (wave & 1) * 32;          // wave col group

    v8f acc0 = {}; v8f acc1 = {};

    const float* Xrow = X + ((size_t)b * SEQ + s0) * DIN;

    for (int k0 = 0; k0 < DIN; k0 += 32) {
        __syncthreads();
        // X tile 64x32: 512 float4 loads, 2 per thread, -> ds_store_b64
        #pragma unroll
        for (int it = 0; it < 2; ++it) {
            int idx = tid + it * 256;          // 0..511
            int r   = idx >> 3;                // 0..63
            int c4  = (idx & 7) * 4;           // 0,4,..,28
            float4 x = *(const float4*)&Xrow[(size_t)r * DIN + k0 + c4];
            v4h h; h.x = (_Float16)x.x; h.y = (_Float16)x.y;
            h.z = (_Float16)x.z; h.w = (_Float16)x.w;
            *(v4h*)&Xs[r * 40 + c4] = h;
        }
        // W tile 32x64 -> Wts[n][k]: 4 coalesced b32 loads per thread -> b64 LDS store
        #pragma unroll
        for (int it = 0; it < 2; ++it) {
            int idx = tid + it * 256;          // 0..511
            int n   = idx & 63;
            int k4  = (idx >> 6) * 4;          // 0,4,..,28
            const float* wp = &W[(size_t)(k0 + k4) * DNEW + n0 + n];
            v4h h;
            h.x = (_Float16)wp[0];
            h.y = (_Float16)wp[DNEW];
            h.z = (_Float16)wp[2 * DNEW];
            h.w = (_Float16)wp[3 * DNEW];
            *(v4h*)&Wts[n * 40 + k4] = h;
        }
        __syncthreads();

        const int ml    = m0 + (lane & 15);
        const int kb    = (lane >> 4) * 8;     // unused here; A also takes K 0..15 / 16..31
        const int kbase = (lane >> 4) * 16;    // B: K 0..15 / 16..31
        V16 a;
        a.h[0] = *(const v8h*)&Xs[ml * 40 + kbase];
        a.h[1] = *(const v8h*)&Xs[ml * 40 + kbase + 8];
        {
            int n = n0w + (lane & 15);
            V16 bf;
            bf.h[0] = *(const v8h*)&Wts[n * 40 + kbase];
            bf.h[1] = *(const v8h*)&Wts[n * 40 + kbase + 8];
            acc0 = __builtin_amdgcn_wmma_f32_16x16x32_f16(false, a.v, false, bf.v,
                                                          (short)0, acc0, false, false);
        }
        {
            int n = n0w + 16 + (lane & 15);
            V16 bf;
            bf.h[0] = *(const v8h*)&Wts[n * 40 + kbase];
            bf.h[1] = *(const v8h*)&Wts[n * 40 + kbase + 8];
            acc1 = __builtin_amdgcn_wmma_f32_16x16x32_f16(false, a.v, false, bf.v,
                                                          (short)0, acc1, false, false);
        }
    }

    // ---- epilogue: bias + scale -> f16, stage in LDS in OUTPUT layout ----
    __syncthreads();
    // convert with bias/scale first (branch-free)
    V8H hv[2];
    #pragma unroll
    for (int nt = 0; nt < 2; ++nt) {
        v8f acc = nt ? acc1 : acc0;
        float bn = bias[n0 + n0w + nt * 16 + (lane & 15)];
        #pragma unroll
        for (int v = 0; v < 8; ++v)
            hv[nt].e[v] = (_Float16)((acc[v] + bn) * scale);
    }
    const int mbase = m0 + ((lane >> 4) * 8);          // first of 8 contiguous rows
    if (z < 2) {                                       // [s][e]: scatter b16 per row
        #pragma unroll
        for (int nt = 0; nt < 2; ++nt) {
            int n = n0w + nt * 16 + (lane & 15);
            #pragma unroll
            for (int v = 0; v < 8; ++v)
                Cs[(mbase + v) * 72 + n] = hv[nt].e[v];
        }
    } else {                                           // [e][s]: 8 rows contiguous -> b128
        #pragma unroll
        for (int nt = 0; nt < 2; ++nt) {
            int n = n0w + nt * 16 + (lane & 15);
            *(v8h*)&Cs[n * 72 + mbase] = hv[nt].v;
        }
    }
    __syncthreads();
    // cooperative coalesced b128 stores; z is block-uniform -> one branch
    if (z < 2) {
        _Float16* outp = (z == 0) ? Q : K;
        #pragma unroll
        for (int it = 0; it < 2; ++it) {
            int idx = tid + it * 256;                  // 0..511
            int r   = idx >> 3;                        // 0..63
            int c8  = (idx & 7) * 8;                   // 0,8,..,56
            *(v8h*)&outp[((size_t)b * SEQ + s0 + r) * DNEW + n0 + c8] =
                *(const v8h*)&Cs[r * 72 + c8];
        }
    } else {
        #pragma unroll
        for (int it = 0; it < 2; ++it) {
            int idx = tid + it * 256;
            int r   = idx >> 3;
            int c8  = (idx & 7) * 8;
            *(v8h*)&Vt[((size_t)b * DNEW + n0 + r) * SEQ + s0 + c8] =
                *(const v8h*)&Cs[r * 72 + c8];
        }
    }
}

// -----------------------------------------------------------------------------
// Kernel 2: flash attention. One workgroup = 32 q-rows, 8 waves:
//   wr = wave>>2 in {0,1}: 16 q-rows each; wc = wave&3 in {0..3}: 128 e-cols each.
// Per 32-key tile: 8 WMMAs for the wave's e-slice of Q*K^T, partials summed via
// LDS (float4 reads), online softmax in fp32 (exp2; scales folded into Q),
// then 8 WMMAs of P*V into fp32 accumulators. Next tile prefetched.
// -----------------------------------------------------------------------------
__launch_bounds__(256)
__global__ void attn_kernel(const _Float16* __restrict__ Q,
                            const _Float16* __restrict__ K,
                            const _Float16* __restrict__ Vt,
                            float* __restrict__ out)
{
    const int b  = blockIdx.x >> 6;            // 64 q-tiles per batch
    const int q0 = (blockIdx.x & 63) * 32;

    const int tid  = threadIdx.x;
    const int lane = tid & 31;
    const int wave = tid >> 5;
    const int wr   = wave >> 2;                // 0..1
    const int wc   = wave & 3;                 // 0..3

    __shared__ float S_lds[4][32][36];         // per-wc partial scores, padded

    const _Float16* Qb = Q  + (size_t)b * SEQ * DNEW;
    const _Float16* Kb = K  + (size_t)b * SEQ * DNEW;
    const _Float16* Vb = Vt + (size_t)b * DNEW * SEQ;

    const int mrow  = q0 + wr * 16 + (lane & 15);
    const int kb    = (lane >> 4) * 8;
    const int kbase = (lane >> 4) * 16;

    // preload Q fragments for this wave's 128-wide e-slice (4 chunks of 32)
    V16 qfrag[4];
    #pragma unroll
    for (int j = 0; j < 4; ++j) {
        const _Float16* p = Qb + (size_t)mrow * DNEW + wc * 128 + j * 32;
        qfrag[j].h[0] = *(const v8h*)(p + kbase);
        qfrag[j].h[1] = *(const v8h*)(p + kbase + 8);
    }

    v8f o[8];
    #pragma unroll
    for (int i = 0; i < 8; ++i) o[i] = (v8f){};
    float m_run = -1e30f;
    float l_run = 0.f;

    for (int key0 = 0; key0 < SEQ; key0 += 32) {
        // prefetch next key tile's K/V rows (global_prefetch_b8)
        if (key0 + 32 < SEQ) {
            __builtin_prefetch(Kb + (size_t)(key0 + 32 + (lane & 15)) * DNEW
                                  + wc * 128 + kbase, 0, 3);
            __builtin_prefetch(Vb + (size_t)(wc * 128 + (lane & 15)) * SEQ
                                  + key0 + 32 + kbase, 0, 3);
        }

        // ---- partial S = Q_slice * K_slice^T (16 x 32 keys) ----
        v8f sp0 = {}, sp1 = {};
        #pragma unroll
        for (int j = 0; j < 4; ++j) {
            int ec = wc * 128 + j * 32;
            {
                int key = key0 + (lane & 15);
                const _Float16* p = Kb + (size_t)key * DNEW + ec + kbase;
                V16 bf; bf.h[0] = *(const v8h*)p; bf.h[1] = *(const v8h*)(p + 8);
                sp0 = __builtin_amdgcn_wmma_f32_16x16x32_f16(false, qfrag[j].v, false, bf.v,
                                                             (short)0, sp0, false, false);
            }
            {
                int key = key0 + 16 + (lane & 15);
                const _Float16* p = Kb + (size_t)key * DNEW + ec + kbase;
                V16 bf; bf.h[0] = *(const v8h*)p; bf.h[1] = *(const v8h*)(p + 8);
                sp1 = __builtin_amdgcn_wmma_f32_16x16x32_f16(false, qfrag[j].v, false, bf.v,
                                                             (short)0, sp1, false, false);
            }
        }
        __syncthreads();   // previous iteration's consumers are done with S_lds
        #pragma unroll
        for (int v = 0; v < 8; ++v) {
            int r = wr * 16 + (lane >> 4) * 8 + v;
            S_lds[wc][r][lane & 15]        = sp0[v];
            S_lds[wc][r][16 + (lane & 15)] = sp1[v];
        }
        __syncthreads();

        // ---- online softmax: lane handles row (lane&15), its 16 A-layout keys ----
        const int rloc = wr * 16 + (lane & 15);
        float sv[16];
        #pragma unroll
        for (int h = 0; h < 16; ++h) sv[h] = 0.f;
        #pragma unroll
        for (int c = 0; c < 4; ++c) {
            const float4 f0 = *(const float4*)&S_lds[c][rloc][kb];
            const float4 f1 = *(const float4*)&S_lds[c][rloc][kb + 4];
            const float4 f2 = *(const float4*)&S_lds[c][rloc][kb + 16];
            const float4 f3 = *(const float4*)&S_lds[c][rloc][kb + 20];
            sv[0] += f0.x;  sv[1] += f0.y;  sv[2]  += f0.z;  sv[3]  += f0.w;
            sv[4] += f1.x;  sv[5] += f1.y;  sv[6]  += f1.z;  sv[7]  += f1.w;
            sv[8] += f2.x;  sv[9] += f2.y;  sv[10] += f2.z;  sv[11] += f2.w;
            sv[12]+= f3.x;  sv[13]+= f3.y;  sv[14] += f3.z;  sv[15] += f3.w;
        }
        float mx = sv[0];
        #pragma unroll
        for (int h = 1; h < 16; ++h) mx = fmaxf(mx, sv[h]);
        mx = fmaxf(mx, xor16(mx));                       // other 16 keys live in lane^16
        float m_new = fmaxf(m_run, mx);
        float sum = 0.f;
        V16 pfrag;
        #pragma unroll
        for (int h = 0; h < 16; ++h) {
            float p = __builtin_exp2f(sv[h] - m_new);    // log2(e) folded into Q
            sum += p;
            pfrag.v[h] = (_Float16)p;
        }
        sum += xor16(sum);
        float alpha = __builtin_exp2f(m_run - m_new);
        l_run = l_run * alpha + sum;
        m_run = m_new;

        // rescale accumulators: alpha for row 8*(lane>>4)+v lives in that lane
        float arow[8];
        #pragma unroll
        for (int v = 0; v < 8; ++v) arow[v] = lane_bcast(alpha, (lane >> 4) * 8 + v);
        #pragma unroll
        for (int i = 0; i < 8; ++i)
            #pragma unroll
            for (int v = 0; v < 8; ++v) o[i][v] *= arow[v];

        // ---- O += P * V over this wave's 128 e-cols (8 subtiles of 16) ----
        #pragma unroll
        for (int nt = 0; nt < 8; ++nt) {
            int e = wc * 128 + nt * 16 + (lane & 15);
            const _Float16* p = Vb + (size_t)e * SEQ + key0 + kb;
            V16 bf; bf.h[0] = *(const v8h*)p; bf.h[1] = *(const v8h*)(p + 16);
            o[nt] = __builtin_amdgcn_wmma_f32_16x16x32_f16(false, pfrag.v, false, bf.v,
                                                           (short)0, o[nt], false, false);
        }
    }

    // ---- finalize: divide by row sum, store fp32 ----
    float linv = 1.0f / l_run;                  // valid for row lane&15
    float lrow[8];
    #pragma unroll
    for (int v = 0; v < 8; ++v) lrow[v] = lane_bcast(linv, (lane >> 4) * 8 + v);
    float* ob = out + (size_t)b * SEQ * DNEW;
    #pragma unroll
    for (int nt = 0; nt < 8; ++nt) {
        int e = wc * 128 + nt * 16 + (lane & 15);
        #pragma unroll
        for (int v = 0; v < 8; ++v) {
            int row = q0 + wr * 16 + (lane >> 4) * 8 + v;
            ob[(size_t)row * DNEW + e] = o[nt][v] * lrow[v];
        }
    }
}

// -----------------------------------------------------------------------------
extern "C" void kernel_launch(void* const* d_in, const int* in_sizes, int n_in,
                              void* d_out, int out_size, void* d_ws, size_t ws_size,
                              hipStream_t stream) {
    (void)in_sizes; (void)n_in; (void)out_size; (void)ws_size;
    const float* X  = (const float*)d_in[0];
    const float* Wq = (const float*)d_in[1];
    const float* bq = (const float*)d_in[2];
    const float* Wk = (const float*)d_in[3];
    const float* bk = (const float*)d_in[4];
    const float* Wv = (const float*)d_in[5];
    const float* bv = (const float*)d_in[6];

    _Float16* ws = (_Float16*)d_ws;
    const size_t MAT = (size_t)NB * SEQ * DNEW;        // 8.39M halves each
    _Float16* Qh = ws;
    _Float16* Kh = ws + MAT;
    _Float16* Vh = ws + 2 * MAT;

    qkv_proj_kernel<<<dim3(NB * (SEQ / 64), DNEW / 64, 3), 256, 0, stream>>>(
        X, Wq, bq, Wk, bk, Wv, bv, Qh, Kh, Vh);
    attn_kernel<<<dim3(NB * (SEQ / 32)), 256, 0, stream>>>(
        Qh, Kh, Vh, (float*)d_out);
}
